// PairEmbed_44736379355620
// MI455X (gfx1250) — matrix operand order repaired and checked by reading, not compiled, not run on hardware
//
#include <hip/hip_runtime.h>
#include <hip/hip_bf16.h>
#include <math.h>

typedef __attribute__((ext_vector_type(16))) _Float16 v16h;
typedef __attribute__((ext_vector_type(8)))  _Float16 v8h;
typedef __attribute__((ext_vector_type(8)))  float    v8f;

#define SPART 128
#define DCH   64
#define NB    64
#define NPAIR 8128              // S*(S-1)/2
#define TILES_PER_BATCH 508     // NPAIR / 16
#define NTILES (NB * TILES_PER_BATCH)
#define FEPS   1e-8f
#define BNEPS  1e-5f

__device__ __forceinline__ float gelu_exact(float v) {
    return 0.5f * v * (1.0f + erff(v * 0.70710678118654752f));
}

// Opaque 32-bit value: defeats LICM so LDS loads stay inside the tile loop
// (keeps register pressure low instead of pinning weight fragments).
__device__ __forceinline__ unsigned opq(unsigned v) {
    asm volatile("" : "+v"(v));
    return v;
}

// A-fragment (16x32 f16): per-lane = two contiguous 8-half runs of the
// row-major matrix: [row][col .. col+7] and [row][col+16 .. col+23].
__device__ __forceinline__ v16h load_afrag(const _Float16* __restrict__ w,
                                           int stride, int row, int col) {
    unsigned o = opq((unsigned)(row * stride + col));
    v8h lo = *(const v8h*)&w[o];
    v8h hi = *(const v8h*)&w[o + 16];
    return __builtin_shufflevector(lo, hi,
        0,1,2,3,4,5,6,7,8,9,10,11,12,13,14,15);
}

__global__ void __launch_bounds__(256)
pairembed_main(const float* __restrict__ x,
               const float* __restrict__ bn0g, const float* __restrict__ bn0b,
               const float* __restrict__ bn0m, const float* __restrict__ bn0v,
               const float* __restrict__ bn1g, const float* __restrict__ bn1b,
               const float* __restrict__ bn1m, const float* __restrict__ bn1v,
               const float* __restrict__ bn2g, const float* __restrict__ bn2b,
               const float* __restrict__ bn2m, const float* __restrict__ bn2v,
               const float* __restrict__ bn3g, const float* __restrict__ bn3b,
               const float* __restrict__ bn3m, const float* __restrict__ bn3v,
               const float* __restrict__ w1, const float* __restrict__ b1,
               const float* __restrict__ w2, const float* __restrict__ b2,
               const float* __restrict__ w3, const float* __restrict__ b3,
               float* __restrict__ out)
{
    // Folded parameters in LDS
    __shared__ __align__(32) _Float16 w1s[DCH * 32];    // s1*w1*s0, zero-padded K=4->32
    __shared__ __align__(32) _Float16 w2s[DCH * DCH];   // s2 * w2, f16 row-major
    __shared__ __align__(32) _Float16 w3s[DCH * DCH];   // s3 * w3, f16 row-major
    __shared__ float b1f[DCH], b2f[DCH], b3f[DCH];
    // Per-wave activation staging in WMMA B-fragment layout:
    // gst[wave][kt(2)][n(16)][klocal(32)] f16
    __shared__ __align__(64) _Float16 gst[8 * 2 * 16 * 32];

    // ---- fold BN into weights/biases (threads 0..63, one output channel each)
    if (threadIdx.x < DCH) {
        const int o = threadIdx.x;
        float s0[4], t0[4];
        #pragma unroll
        for (int k = 0; k < 4; ++k) {
            s0[k] = bn0g[k] / sqrtf(bn0v[k] + BNEPS);
            t0[k] = bn0b[k] - bn0m[k] * s0[k];
        }
        const float s1 = bn1g[o] / sqrtf(bn1v[o] + BNEPS);
        const float t1 = bn1b[o] - bn1m[o] * s1;
        const float s2 = bn2g[o] / sqrtf(bn2v[o] + BNEPS);
        const float t2 = bn2b[o] - bn2m[o] * s2;
        const float s3 = bn3g[o] / sqrtf(bn3v[o] + BNEPS);
        const float t3 = bn3b[o] - bn3m[o] * s3;

        float acc = b1[o];
        #pragma unroll
        for (int k = 0; k < 4; ++k) {
            w1s[o * 32 + k] = (_Float16)(s1 * w1[o * 4 + k] * s0[k]);
            acc += w1[o * 4 + k] * t0[k];
        }
        for (int k = 4; k < 32; ++k) w1s[o * 32 + k] = (_Float16)0.0f;
        b1f[o] = s1 * acc + t1;
        for (int k = 0; k < DCH; ++k) {
            w2s[o * DCH + k] = (_Float16)(s2 * w2[o * DCH + k]);
            w3s[o * DCH + k] = (_Float16)(s3 * w3[o * DCH + k]);
        }
        b2f[o] = s2 * b2[o] + t2;
        b3f[o] = s3 * b3[o] + t3;
    }
    __syncthreads();

    const int wave = threadIdx.x >> 5;
    const int lane = threadIdx.x & 31;
    const int n    = lane & 15;     // token within tile / N index
    const int hi   = lane >> 4;     // half-wave selector

    _Float16* gw = &gst[wave * (2 * 16 * 32)];

    const float TWO_PI = 6.283185307179586f;
    const float PI_F   = 3.141592653589793f;

    // ---- persistent loop over 16-pair tiles (one tile per wave per iteration)
    for (int tile = blockIdx.x * 8 + wave; tile < NTILES; tile += gridDim.x * 8) {
        const int bidx = tile / TILES_PER_BATCH;
        const int pp   = (tile - bidx * TILES_PER_BATCH) * 16 + n;

        // pair index -> (i, j) with pp = i*(i-1)/2 + j, j < i
        int i = (int)((1.0f + sqrtf(1.0f + 8.0f * (float)pp)) * 0.5f);
        while ((i * (i - 1)) / 2 > pp) --i;
        while (((i + 1) * i) / 2 <= pp) ++i;
        const int j = pp - (i * (i - 1)) / 2;

        // ---- pair features
        const float* xb = x + (size_t)bidx * 4 * SPART;
        const float pxi = xb[i],             pyi = xb[SPART + i];
        const float pzi = xb[2 * SPART + i], ei  = xb[3 * SPART + i];
        const float pxj = xb[j],             pyj = xb[SPART + j];
        const float pzj = xb[2 * SPART + j], ej  = xb[3 * SPART + j];

        const float pti  = sqrtf(pxi * pxi + pyi * pyi);
        const float ptj  = sqrtf(pxj * pxj + pyj * pyj);
        const float rapi = 0.5f * logf(1.0f + 2.0f * pzi / fmaxf(ei - pzi, 1e-20f));
        const float rapj = 0.5f * logf(1.0f + 2.0f * pzj / fmaxf(ej - pzj, 1e-20f));
        const float phii = atan2f(pyi, pxi);
        const float phij = atan2f(pyj, pxj);

        float dphi = fmodf(phii - phij + PI_F, TWO_PI);
        if (dphi < 0.0f) dphi += TWO_PI;
        dphi -= PI_F;
        const float drap  = rapi - rapj;
        const float delta = sqrtf(drap * drap + dphi * dphi);
        const float ptmin = fminf(pti, ptj);

        const float f2 = logf(fmaxf(delta, FEPS));                            // lndelta
        const float f0 = logf(fmaxf(ptmin * delta, FEPS));                    // lnkt
        const float f1 = logf(fmaxf(ptmin / fmaxf(pti + ptj, FEPS), FEPS));   // lnz
        const float sx = pxi + pxj, sy = pyi + pyj, sz = pzi + pzj, se = ei + ej;
        const float m2 = se * se - (sx * sx + sy * sy + sz * sz);
        const float f3 = logf(fmaxf(m2, FEPS));                               // lnm2

        // ---- layer 1 (4 -> 64) via WMMA with zero-padded K.
        // Feature B-fragment built in registers: lanes 0-15 hold K=0..7,16..23 of the
        // tile (features live at K=0..3); lanes 16-31 hold K=8..15,24..31 which hit
        // zero columns of w1s, so their content is don't-care.
        v8f a1[4];
        {
            const _Float16 z0 = (_Float16)0.0f;
            v16h bf1 = { (_Float16)f0, (_Float16)f1, (_Float16)f2, (_Float16)f3,
                         z0,z0,z0,z0, z0,z0,z0,z0, z0,z0,z0,z0 };
            #pragma unroll
            for (int mt = 0; mt < 4; ++mt) {
                v16h af = load_afrag(w1s, 32, mt * 16 + n, hi * 8);
                a1[mt] = __builtin_amdgcn_wmma_f32_16x16x32_f16(
                    false, af, false, bf1, (short)0,
                    (v8f){0,0,0,0,0,0,0,0}, false, false);
            }
        }
        // bias + GELU -> stage as B-fragments for layer 2
        #pragma unroll
        for (int mt = 0; mt < 4; ++mt) {
            const int c0 = mt * 16 + hi * 8;   // 8 contiguous channels
            const unsigned bo = opq((unsigned)c0);
            v8h pk;
            #pragma unroll
            for (int r = 0; r < 8; ++r)
                pk[r] = (_Float16)gelu_exact(a1[mt][r] + b1f[bo + r]);
            *(v8h*)&gw[((c0 >> 5) * 16 + n) * 32 + (c0 & 31)] = pk;
        }

        // ---- layer 2: 64x64 GEMM via 8 WMMAs
        v8f a2[4];
        #pragma unroll
        for (int mt = 0; mt < 4; ++mt) a2[mt] = (v8f){0,0,0,0,0,0,0,0};
        #pragma unroll
        for (int kt = 0; kt < 2; ++kt) {
            const _Float16* gb = &gw[(kt * 16 + n) * 32 + hi * 8];
            v16h bfr = __builtin_shufflevector(*(const v8h*)gb, *(const v8h*)(gb + 16),
                0,1,2,3,4,5,6,7,8,9,10,11,12,13,14,15);
            #pragma unroll
            for (int mt = 0; mt < 4; ++mt) {
                v16h af = load_afrag(w2s, DCH, mt * 16 + n, kt * 32 + hi * 8);
                a2[mt] = __builtin_amdgcn_wmma_f32_16x16x32_f16(
                    false, af, false, bfr, (short)0, a2[mt], false, false);
            }
        }
        // bias + GELU -> restage as B-fragments for layer 3
        #pragma unroll
        for (int mt = 0; mt < 4; ++mt) {
            const int c0 = mt * 16 + hi * 8;
            const unsigned bo = opq((unsigned)c0);
            v8h pk;
            #pragma unroll
            for (int r = 0; r < 8; ++r)
                pk[r] = (_Float16)gelu_exact(a2[mt][r] + b2f[bo + r]);
            *(v8h*)&gw[((c0 >> 5) * 16 + n) * 32 + (c0 & 31)] = pk;
        }

        // ---- layer 3: 64x64 GEMM via 8 WMMAs
        v8f a3[4];
        #pragma unroll
        for (int mt = 0; mt < 4; ++mt) a3[mt] = (v8f){0,0,0,0,0,0,0,0};
        #pragma unroll
        for (int kt = 0; kt < 2; ++kt) {
            const _Float16* gb = &gw[(kt * 16 + n) * 32 + hi * 8];
            v16h bfr = __builtin_shufflevector(*(const v8h*)gb, *(const v8h*)(gb + 16),
                0,1,2,3,4,5,6,7,8,9,10,11,12,13,14,15);
            #pragma unroll
            for (int mt = 0; mt < 4; ++mt) {
                v16h af = load_afrag(w3s, DCH, mt * 16 + n, kt * 32 + hi * 8);
                a3[mt] = __builtin_amdgcn_wmma_f32_16x16x32_f16(
                    false, af, false, bfr, (short)0, a3[mt], false, false);
            }
        }

        // ---- final bias (bn3 folded) + symmetric scatter stores (32-bit offsets)
        #pragma unroll
        for (int mt = 0; mt < 4; ++mt) {
            const int c0 = mt * 16 + hi * 8;
            const unsigned bo = opq((unsigned)c0);
            const unsigned cb = ((unsigned)bidx * DCH + (unsigned)c0) << 14;
            const unsigned o_ij = cb + ((unsigned)i << 7) + (unsigned)j;
            const unsigned o_ji = cb + ((unsigned)j << 7) + (unsigned)i;
            #pragma unroll
            for (int r = 0; r < 8; ++r) {
                const float val = a3[mt][r] + b3f[bo + r];
                out[o_ij + ((unsigned)r << 14)] = val;
                out[o_ji + ((unsigned)r << 14)] = val;
            }
        }
    }
}

__global__ void __launch_bounds__(256)
pairembed_diag_zero(float* __restrict__ out)
{
    const int idx = blockIdx.x * blockDim.x + threadIdx.x;   // 0 .. B*D*S-1
    const int s  = idx & (SPART - 1);
    const int bc = idx >> 7;                                  // b*64 + c
    out[((size_t)bc * SPART + s) * SPART + s] = 0.0f;
}

extern "C" void kernel_launch(void* const* d_in, const int* in_sizes, int n_in,
                              void* d_out, int out_size, void* d_ws, size_t ws_size,
                              hipStream_t stream) {
    (void)in_sizes; (void)n_in; (void)d_ws; (void)ws_size; (void)out_size;
    const float* x    = (const float*)d_in[0];
    const float* bn0g = (const float*)d_in[1];
    const float* bn0b = (const float*)d_in[2];
    const float* bn0m = (const float*)d_in[3];
    const float* bn0v = (const float*)d_in[4];
    const float* bn1g = (const float*)d_in[5];
    const float* bn1b = (const float*)d_in[6];
    const float* bn1m = (const float*)d_in[7];
    const float* bn1v = (const float*)d_in[8];
    const float* bn2g = (const float*)d_in[9];
    const float* bn2b = (const float*)d_in[10];
    const float* bn2m = (const float*)d_in[11];
    const float* bn2v = (const float*)d_in[12];
    const float* bn3g = (const float*)d_in[13];
    const float* bn3b = (const float*)d_in[14];
    const float* bn3m = (const float*)d_in[15];
    const float* bn3v = (const float*)d_in[16];
    const float* w1   = (const float*)d_in[17];
    const float* b1   = (const float*)d_in[18];
    const float* w2   = (const float*)d_in[19];
    const float* b2   = (const float*)d_in[20];
    const float* w3   = (const float*)d_in[21];
    const float* b3   = (const float*)d_in[22];
    float* out = (float*)d_out;

    // Zero the diagonal (B*D*S = 524288 elements)
    pairembed_diag_zero<<<(NB * DCH * SPART) / 256, 256, 0, stream>>>(out);

    // Main fused kernel: 1024 blocks x 256 threads (8 waves), persistent tiles
    pairembed_main<<<1024, 256, 0, stream>>>(
        x, bn0g, bn0b, bn0m, bn0v, bn1g, bn1b, bn1m, bn1v,
        bn2g, bn2b, bn2m, bn2v, bn3g, bn3b, bn3m, bn3v,
        w1, b1, w2, b2, w3, b3, out);
}
